// GCN2_LC_NL_FC1_22385369546848
// MI455X (gfx1250) — matrix operand order repaired and checked by reading, not compiled, not run on hardware
//
#include <hip/hip_runtime.h>

// ---------------------------------------------------------------------------
// GCN forward for MI455X (gfx1250): bf16 WMMA GEMMs + run-compressed SpMM.
// ---------------------------------------------------------------------------

typedef __attribute__((ext_vector_type(16))) __bf16 v16bf;
typedef __attribute__((ext_vector_type(8)))  __bf16 v8bf;
typedef __attribute__((ext_vector_type(8)))  float  v8f;

#define N_NODES 100000
#define N_EDGES 1600000

__device__ __forceinline__ unsigned short f32_to_bf16(float f) {
  unsigned int u = __float_as_uint(f);
  u += 0x7FFFu + ((u >> 16) & 1u);           // round-to-nearest-even
  return (unsigned short)(u >> 16);
}

// ---------------------------------------------------------------------------
// Zero-fill (agg buffers must start at 0 for scatter-add).
// ---------------------------------------------------------------------------
__global__ void zero_f32(float* __restrict__ p, int n) {
  int i = blockIdx.x * blockDim.x + threadIdx.x;
  int stride = gridDim.x * blockDim.x;
  for (; i < n; i += stride) p[i] = 0.f;
}

// ---------------------------------------------------------------------------
// Pre-swizzle a row-major f32 weight [K, Nreal] into bf16 WMMA B-fragments.
// Fragment layout (contiguous k order, same as the A loads): for tile (kt, nt), lane L,
// half-slot j:  value = B[kt*32 + (L>=16 ? 16 : 0) + j][nt*16 + (L&15)].
// Linear index = ((kt*NT + nt)*32 + L)*16 + j.  Columns >= Nreal are zero.
// ---------------------------------------------------------------------------
__global__ void make_bfrag(const float* __restrict__ w, unsigned short* __restrict__ frag,
                           int K, int Nreal, int NT) {
  int tid = blockIdx.x * blockDim.x + threadIdx.x;
  int total = (K / 32) * NT * 512;
  if (tid >= total) return;
  int j  = tid & 15;
  int L  = (tid >> 4) & 31;
  int t  = tid >> 9;
  int nt = t % NT;
  int kt = t / NT;
  int k  = kt * 32 + ((L >> 4) ? 16 : 0) + j;
  int n  = nt * 16 + (L & 15);
  float f = (n < Nreal) ? w[(size_t)k * Nreal + n] : 0.f;
  frag[tid] = f32_to_bf16(f);
}

// ---------------------------------------------------------------------------
// WMMA GEMM:  C[16 rows per block, N] = A[16, K] @ Bfrag[K, N]
// A is assembled on the fly from up to two f32 source segments, each with an
// optional bias+ReLU (fuses the GCN layer epilogue into the next GEMM's load,
// and fuses the final concat([x2, x1])).  A tile staged in LDS as bf16.
// One wave per 16x16 C tile; NT waves per block; grid.x = M/16.
// ---------------------------------------------------------------------------
template <int K, int NT>
__global__ __launch_bounds__(NT * 32)
void gemm_wmma(const float* __restrict__ A0, int ld0, const float* __restrict__ bias0, int K0,
               const float* __restrict__ A1, int ld1, const float* __restrict__ bias1,
               const unsigned short* __restrict__ bfrag,
               float* __restrict__ C, int ldc, int ncols) {
  __shared__ alignas(16) unsigned short As[16 * K];   // 16 rows x K bf16

  const int tid = threadIdx.x;
  const int m0  = blockIdx.x * 16;                    // M = 100000 = 6250*16, no tail

  // Cooperative LDS fill with fused segment select + bias + ReLU + bf16 cvt.
  for (int idx = tid; idx < 16 * K; idx += NT * 32) {
    int m = idx / K;
    int k = idx - m * K;
    int row = m0 + m;
    float f;
    if (k < K0) {
      f = A0[(size_t)row * ld0 + k];
      if (bias0) f = fmaxf(f + bias0[k], 0.f);
    } else {
      int kk = k - K0;
      f = fmaxf(A1[(size_t)row * ld1 + kk] + bias1[kk], 0.f);
    }
    As[idx] = f32_to_bf16(f);
  }
  __syncthreads();

  const int lane = tid & 31;
  const int nt   = tid >> 5;
  const int m    = lane & 15;
  const int hs   = (lane >> 4) * 16;  // lanes 16-31 hold the +16 K-offset half

  v8f acc = {0.f, 0.f, 0.f, 0.f, 0.f, 0.f, 0.f, 0.f};

#pragma unroll
  for (int kt = 0; kt < K / 32; ++kt) {
    // A fragment: lane holds K = kt*32 + {hs..hs+15} of row m.
    const int kb = kt * 32 + hs;
    v8bf alo = *reinterpret_cast<const v8bf*>(&As[m * K + kb]);        // ds_load_b128
    v8bf ahi = *reinterpret_cast<const v8bf*>(&As[m * K + kb + 8]);    // ds_load_b128
    v16bf a = __builtin_shufflevector(alo, ahi, 0, 1, 2, 3, 4, 5, 6, 7,
                                      8, 9, 10, 11, 12, 13, 14, 15);
    // B fragment: pre-swizzled, one aligned 32B load per lane.
    v16bf b = *reinterpret_cast<const v16bf*>(
        bfrag + ((size_t)(kt * NT + nt) * 512 + lane * 16));
    acc = __builtin_amdgcn_wmma_f32_16x16x32_bf16(
        false, a, false, b, (short)0, acc, false, false);
  }

  // C layout: VGPR v -> M = v + (lane>=16 ? 8 : 0), N = nt*16 + (lane&15).
  const int col = nt * 16 + (lane & 15);
  if (col < ncols) {
    const int rb = m0 + (lane >> 4) * 8;
#pragma unroll
    for (int v = 0; v < 8; ++v)
      C[(size_t)(rb + v) * ldc + col] = acc[v];
  }
}

// ---------------------------------------------------------------------------
// SpMM scatter:  agg[dst] += sup[src] * val   over sorted-by-dst edge list.
// One wave owns a contiguous edge chunk; lanes split the feature dim
// (VEC floats/lane, NF = 32*VEC).  Runs of equal dst accumulate in registers;
// atomics only at run boundaries (~16x atomic reduction at avg degree 16).
// Edge data loaded coalesced 32-at-a-time and broadcast via __shfl.
// ---------------------------------------------------------------------------
template <int VEC>
__global__ void spmm_scatter(const int* __restrict__ erow, const int* __restrict__ ecol,
                             const float* __restrict__ eval,
                             const float* __restrict__ sup, float* __restrict__ agg,
                             int nEdges, int edgesPerWave) {
  const int NF   = VEC * 32;
  const int lane = threadIdx.x & 31;
  const int wave = threadIdx.x >> 5;
  const int chunk = blockIdx.x * (blockDim.x >> 5) + wave;
  long e0 = (long)chunk * edgesPerWave;
  if (e0 >= nEdges) return;
  long e1 = e0 + edgesPerWave;
  if (e1 > nEdges) e1 = nEdges;

  float acc[VEC];
#pragma unroll
  for (int k = 0; k < VEC; ++k) acc[k] = 0.f;
  int cur = -1;

  for (long base = e0; base < e1; base += 32) {
    int r = 0, c = 0;
    float v = 0.f;
    long ee = base + lane;
    if (ee < e1) { r = erow[ee]; c = ecol[ee]; v = eval[ee]; }
    int cnt = (int)(((e1 - base) < 32) ? (e1 - base) : 32);
    for (int i = 0; i < cnt; ++i) {
      int   dr = __shfl(r, i, 32);
      int   dc = __shfl(c, i, 32);
      float dv = __shfl(v, i, 32);
      if (dr != cur) {
        if (cur >= 0) {
#pragma unroll
          for (int k = 0; k < VEC; ++k)
            atomicAdd(&agg[(size_t)cur * NF + lane * VEC + k], acc[k]);
        }
        cur = dr;
#pragma unroll
        for (int k = 0; k < VEC; ++k) acc[k] = 0.f;
      }
      const float* sp = sup + (size_t)dc * NF + lane * VEC;   // coalesced row gather
#pragma unroll
      for (int k = 0; k < VEC; ++k) acc[k] = fmaf(sp[k], dv, acc[k]);
    }
  }
  if (cur >= 0) {
#pragma unroll
    for (int k = 0; k < VEC; ++k)
      atomicAdd(&agg[(size_t)cur * NF + lane * VEC + k], acc[k]);
  }
}

// ---------------------------------------------------------------------------
// Row-wise log_softmax over 40 classes (+ final bias).  One wave32 per row.
// ---------------------------------------------------------------------------
__global__ void logsoftmax40(const float* __restrict__ logits, const float* __restrict__ bl,
                             float* __restrict__ out, int M) {
  int gw   = (blockIdx.x * blockDim.x + threadIdx.x) >> 5;   // row
  int lane = threadIdx.x & 31;
  if (gw >= M) return;
  const float* rp = logits + (size_t)gw * 40;
  float v1 = rp[lane] + bl[lane];                            // lanes 0..31 valid
  bool  h2 = lane < 8;
  float v2 = h2 ? (rp[lane + 32] + bl[lane + 32]) : -INFINITY;
  float m = fmaxf(v1, v2);
#pragma unroll
  for (int mask = 16; mask >= 1; mask >>= 1) m = fmaxf(m, __shfl_xor(m, mask, 32));
  float s = expf(v1 - m) + (h2 ? expf(v2 - m) : 0.f);
#pragma unroll
  for (int mask = 16; mask >= 1; mask >>= 1) s += __shfl_xor(s, mask, 32);
  float lse = m + logf(s);
  out[(size_t)gw * 40 + lane] = v1 - lse;
  if (h2) out[(size_t)gw * 40 + lane + 32] = v2 - lse;
}

// ---------------------------------------------------------------------------
extern "C" void kernel_launch(void* const* d_in, const int* in_sizes, int n_in,
                              void* d_out, int out_size, void* d_ws, size_t ws_size,
                              hipStream_t stream) {
  const float* x    = (const float*)d_in[0];
  const int*   erow = (const int*)d_in[1];
  const int*   ecol = (const int*)d_in[2];
  const float* ev   = (const float*)d_in[3];
  const float* w1   = (const float*)d_in[4];
  const float* b1   = (const float*)d_in[5];
  const float* w2   = (const float*)d_in[6];
  const float* b2   = (const float*)d_in[7];
  const float* wl   = (const float*)d_in[8];
  const float* bl   = (const float*)d_in[9];
  float* out = (float*)d_out;

  // Workspace layout (bytes).  Region 0 (0..51.2MB) is time-multiplexed:
  //   phase 1: support1 [100000,128]
  //   phase 2: support2 [100000,64] @ +0   |  agg2 [100000,64] @ +25.6MB
  //   phase 3: logits   [100000,40] @ +0   (agg2 still live, disjoint)
  // Region 1 (51.2..102.4MB): agg1 [100000,128]   (live until final GEMM)
  // Region 2: bf16 weight fragments.
  char* ws = (char*)d_ws;
  float* support1 = (float*)(ws);
  float* support2 = (float*)(ws);
  float* agg2     = (float*)(ws + 25600000);
  float* logits   = (float*)(ws);
  float* agg1     = (float*)(ws + 51200000);
  unsigned short* frag1 = (unsigned short*)(ws + 102400000);   // 256*128
  unsigned short* frag2 = frag1 + 256 * 128;                   // 128*64
  unsigned short* frag3 = frag2 + 128 * 64;                    // 192*48

  // Weight pre-swizzle into WMMA B-fragment order.
  make_bfrag<<<(256 * 128 + 255) / 256, 256, 0, stream>>>(w1, frag1, 256, 128, 8);
  make_bfrag<<<(128 * 64 + 255) / 256, 256, 0, stream>>>(w2, frag2, 128, 64, 4);
  make_bfrag<<<(192 * 48 + 255) / 256, 256, 0, stream>>>(wl, frag3, 192, 40, 3);

  zero_f32<<<2048, 256, 0, stream>>>(agg1, N_NODES * 128);

  // Layer 1 GEMM: support1 = x @ w1            (K=256, N=128 -> 8 waves/block)
  gemm_wmma<256, 8><<<N_NODES / 16, 8 * 32, 0, stream>>>(
      x, 256, nullptr, 256, nullptr, 0, nullptr, frag1, support1, 128, 128);

  // Layer 1 SpMM: agg1 = scatter_add(support1[col] * val -> row)
  const int EPW = 512;
  int nchunks = (N_EDGES + EPW - 1) / EPW;
  spmm_scatter<4><<<(nchunks + 3) / 4, 128, 0, stream>>>(erow, ecol, ev, support1, agg1,
                                                         N_EDGES, EPW);

  zero_f32<<<2048, 256, 0, stream>>>(agg2, N_NODES * 64);

  // Layer 2 GEMM: support2 = relu(agg1 + b1) @ w2   (bias+ReLU fused into A load)
  gemm_wmma<128, 4><<<N_NODES / 16, 4 * 32, 0, stream>>>(
      agg1, 128, b1, 128, nullptr, 0, nullptr, frag2, support2, 64, 64);

  // Layer 2 SpMM
  spmm_scatter<2><<<(nchunks + 3) / 4, 128, 0, stream>>>(erow, ecol, ev, support2, agg2,
                                                         N_EDGES, EPW);

  // Final GEMM: logits = [relu(agg2+b2) ‖ relu(agg1+b1)] @ wl   (concat fused)
  gemm_wmma<192, 3><<<N_NODES / 16, 3 * 32, 0, stream>>>(
      agg2, 64, b2, 64, agg1, 128, b1, frag3, logits, 40, 40);

  // log_softmax(logits + bl)
  logsoftmax40<<<(N_NODES + 7) / 8, 256, 0, stream>>>(logits, bl, out, N_NODES);
}
